// PaiNNBlock_22333829940094
// MI455X (gfx1250) — matrix operand order repaired and checked by reading, not compiled, not run on hardware
//
#include <hip/hip_runtime.h>
#include <hip/hip_bf16.h>
#include <math.h>

typedef __attribute__((ext_vector_type(16))) _Float16 v16h;
typedef __attribute__((ext_vector_type(8)))  _Float16 v8h;
typedef __attribute__((ext_vector_type(8)))  float    v8f;

#define BB  2
#define NN  256
#define NH  192
#define DD  128
#define RR  20
#define TD  384   // 3*D

// ---------------------------------------------------------------------------
// Kernel 1: s_all assembly + rmsnorm(norm1) + GLU(W_i1) + W_i2  -> x (B,N,384)
// One block per (b,i), 128 threads (thread = channel d).
// ---------------------------------------------------------------------------
__global__ __launch_bounds__(128)
void painn_prep_kernel(const float* __restrict__ s_heavy,
                       const int*   __restrict__ is_hydrogen,
                       const float* __restrict__ h_emb,
                       const float* __restrict__ norm1_w,
                       const float* __restrict__ W_i1, const float* __restrict__ b_i1,
                       const float* __restrict__ W_i2, const float* __restrict__ b_i2,
                       float* __restrict__ s_all,
                       float* __restrict__ x)
{
    const int bi = blockIdx.x;
    const int b  = bi / NN, i = bi % NN;
    const int d  = threadIdx.x;

    __shared__ float red[DD];
    __shared__ float sh_rn[DD];
    __shared__ float sh_h[DD];

    const bool isH = (is_hydrogen[bi] != 0);
    float s;
    if (isH)               s = h_emb[d];
    else if (i < NH)       s = s_heavy[((size_t)b * NH + i) * DD + d];
    else                   s = 0.0f;
    s_all[(size_t)bi * DD + d] = s;

    // rmsnorm
    red[d] = s * s;
    __syncthreads();
    for (int o = 64; o > 0; o >>= 1) { if (d < o) red[d] += red[d + o]; __syncthreads(); }
    const float var = red[0] * (1.0f / DD);
    const float rn  = s * rsqrtf(var + 1e-6f) * (1.0f + norm1_w[d]);
    sh_rn[d] = rn;
    __syncthreads();

    // o = rn @ W_i1 + b_i1   (two outputs per thread: d and d+128)
    float a0 = b_i1[d], g0 = b_i1[d + DD];
    #pragma unroll 4
    for (int k = 0; k < DD; ++k) {
        const float r = sh_rn[k];
        a0 = fmaf(r, W_i1[k * 2 * DD + d],      a0);
        g0 = fmaf(r, W_i1[k * 2 * DD + d + DD], g0);
    }
    const float h = (a0 / (1.0f + expf(-a0))) * (1.0f / (1.0f + expf(-g0)));
    sh_h[d] = h;
    __syncthreads();

    // x = h @ W_i2 + b_i2   (three outputs per thread)
    float x0 = b_i2[d], x1 = b_i2[d + DD], x2 = b_i2[d + 2 * DD];
    #pragma unroll 4
    for (int k = 0; k < DD; ++k) {
        const float hh = sh_h[k];
        x0 = fmaf(hh, W_i2[k * TD + d],          x0);
        x1 = fmaf(hh, W_i2[k * TD + d + DD],     x1);
        x2 = fmaf(hh, W_i2[k * TD + d + 2 * DD], x2);
    }
    float* xo = x + (size_t)bi * TD;
    xo[d] = x0; xo[d + DD] = x1; xo[d + 2 * DD] = x2;
}

// ---------------------------------------------------------------------------
// Kernel 2: fused interaction, templated on channel-third G (compile-time).
// grid = B*N per instantiation; block = 256 threads = 8 wave32s.
//   G==0: dq channels   [0,128)   -> q = s_all + sum_j
//   G==1: dmuR channels [128,256) -> dmu1[b,i,r,d] = sum_j val * dir[b,i,j,r]
//   G==2: dmumu chans   [256,384) -> dmu2[b,i,r,d] = sum_j val * v_all[b,j,r,d]
// rbf (K 20->32 zero-padded, fp16) staged in LDS once. x*mask and v_all tiles
// are REGISTER double-buffered: tile jt+1's global_load_b128s are issued
// before the WMMA/reduction of tile jt, so HBM/L2 latency overlaps compute.
// Inner loop: ds_store (staged regs) -> barrier -> prefetch regs ->
// ds_load A-frag -> V_WMMA_F32_16X16X32_F16 (bias via C) -> LDS-fed FMA
// reduction -> barrier.
// ---------------------------------------------------------------------------
template <int G>
__global__ __launch_bounds__(256)
void painn_interaction_kernel(const float* __restrict__ rbf,
                              const float* __restrict__ dir_ij,
                              const float* __restrict__ mask_ij,
                              const float* __restrict__ v_all,
                              const float* __restrict__ W_filter,
                              const float* __restrict__ b_filter,
                              const float* __restrict__ x,
                              const float* __restrict__ s_all,
                              float* __restrict__ q,
                              float* __restrict__ dmu1,
                              float* __restrict__ dmu2)
{
    const int bi   = blockIdx.x;            // b*N + i
    const int b    = bi / NN;
    const int tid  = threadIdx.x;
    const int wave = tid >> 5;
    const int lane = tid & 31;
    const int nloc = lane & 15;             // column within 16-wide tile
    const int hi   = lane >> 4;             // K-half / row-half select
    const int wcol = wave * 16 + nloc;      // channel within this third [0,128)
    const int col  = G * 128 + wcol;        // channel in [0,384)

    __shared__ _Float16 sh_a[NN * 32];                       // 16 KB padded rbf
    __shared__ float    sh_x[16 * 128];                      //  8 KB x*mask tile
    __shared__ float    sh_dir[(G == 1) ? NN * 3 : 1];       //  3 KB (G==1)
    __shared__ float    sh_v[(G == 2) ? 16 * TD : 1];        // 24 KB (G==2)

    // ---- stage zero-padded fp16 rbf for all 256 rows (one row per thread) ---
    {
        const float* arow = rbf + ((size_t)bi * NN + tid) * RR;
        _Float16* dst = &sh_a[tid * 32];
        #pragma unroll
        for (int k = 0; k < RR; ++k) dst[k] = (_Float16)arow[k];
        #pragma unroll
        for (int k = RR; k < 32; ++k) dst[k] = (_Float16)0.0f;
    }
    if (G == 1)
        for (int t = tid; t < NN * 3; t += 256)
            sh_dir[t] = dir_ij[(size_t)bi * NN * 3 + t];

    // ---- B fragment: W_filter[k][col], branch-free zero pad -----------------
    v16h bfrag;
    #pragma unroll
    for (int kk = 0; kk < 16; ++kk) {
        const int k  = (kk < 8) ? (hi * 8 + kk) : (16 + hi * 8 + (kk - 8));
        const int kc = (k < RR) ? k : (RR - 1);               // clamp address
        const float wv = W_filter[kc * TD + col];             // unconditional load
        bfrag[kk] = (k < RR) ? (_Float16)wv : (_Float16)0.0f; // select, no branch
    }
    v8f cfrag;
    {
        const float bias = b_filter[col];
        #pragma unroll
        for (int v = 0; v < 8; ++v) cfrag[v] = bias;
    }

    float acc_dq = 0.0f;
    float acc0 = 0.0f, acc1 = 0.0f, acc2 = 0.0f;

    const int xr = tid >> 4;            // staging row   0..15
    const int xc = (tid & 15) * 8;      // staging col   0..120
    const float* xsrc  = x + ((size_t)b * NN + xr) * TD + G * 128 + xc;
    const float* vsrc  = v_all + ((size_t)b * NN) * TD + tid * 24;
    const float* msrc  = mask_ij + (size_t)bi * NN + xr;

    // ---- prologue: prefetch tile 0 into registers ---------------------------
    float  rm  = msrc[0];
    float4 rx0 = *(const float4*)(xsrc);
    float4 rx1 = *(const float4*)(xsrc + 4);
    float4 rv[6];
    if (G == 2) {
        #pragma unroll
        for (int u = 0; u < 6; ++u) rv[u] = *(const float4*)(vsrc + u * 4);
    }

    __syncthreads();

    for (int jt = 0; jt < NN / 16; ++jt) {
        // ---- commit staged registers to LDS --------------------------------
        {
            float* dx = &sh_x[xr * 128 + xc];
            dx[0] = rx0.x * rm; dx[1] = rx0.y * rm; dx[2] = rx0.z * rm; dx[3] = rx0.w * rm;
            dx[4] = rx1.x * rm; dx[5] = rx1.y * rm; dx[6] = rx1.z * rm; dx[7] = rx1.w * rm;
        }
        if (G == 2) {
            float* dv = &sh_v[tid * 24];
            #pragma unroll
            for (int u = 0; u < 6; ++u) *(float4*)(dv + u * 4) = rv[u];
        }
        __syncthreads();

        // ---- issue next tile's global loads (overlap with WMMA below) ------
        if (jt + 1 < NN / 16) {
            const float* xs = xsrc + (size_t)(jt + 1) * 16 * TD;
            __builtin_prefetch(xs + 16 * TD, 0, 1);           // global_prefetch
            rm  = msrc[(jt + 1) * 16];
            rx0 = *(const float4*)(xs);
            rx1 = *(const float4*)(xs + 4);
            if (G == 2) {
                const float* vs = vsrc + (size_t)(jt + 1) * 16 * TD;
                #pragma unroll
                for (int u = 0; u < 6; ++u) rv[u] = *(const float4*)(vs + u * 4);
            }
        }

        // ---- A fragment: two unconditional 16B LDS reads per lane ----------
        const _Float16* ap = &sh_a[(jt * 16 + nloc) * 32 + hi * 8];
        const v8h a_lo = *(const v8h*)(ap);        // K = hi*8 + 0..7
        const v8h a_hi = *(const v8h*)(ap + 16);   // K = 16 + hi*8 + 0..7
        v16h afrag;
        #pragma unroll
        for (int u = 0; u < 8; ++u) { afrag[u] = a_lo[u]; afrag[u + 8] = a_hi[u]; }

        const v8f dacc = __builtin_amdgcn_wmma_f32_16x16x32_f16(
            /*neg_a=*/false, afrag, /*neg_b=*/false, bfrag,
            /*c_mod=*/(short)0, cfrag, /*reuse_a=*/false, /*reuse_b=*/false);

        // ---- elementwise scale + reduction (pure LDS/reg, compile-time G) --
        #pragma unroll
        for (int v = 0; v < 8; ++v) {
            const int rowl = v + hi * 8;                      // row within tile
            const float val = dacc[v] * sh_x[rowl * 128 + wcol];
            if (G == 0) {
                acc_dq += val;
            } else if (G == 1) {
                const int j = jt * 16 + rowl;
                acc0 = fmaf(val, sh_dir[j * 3 + 0], acc0);
                acc1 = fmaf(val, sh_dir[j * 3 + 1], acc1);
                acc2 = fmaf(val, sh_dir[j * 3 + 2], acc2);
            } else {
                acc0 = fmaf(val, sh_v[rowl * TD + 0 * DD + wcol], acc0);
                acc1 = fmaf(val, sh_v[rowl * TD + 1 * DD + wcol], acc1);
                acc2 = fmaf(val, sh_v[rowl * TD + 2 * DD + wcol], acc2);
            }
        }
        __syncthreads();
    }

    // lanes l and l^16 hold the two row-halves of the same column: merge.
    if (G == 0) {
        acc_dq += __shfl_xor(acc_dq, 16, 32);
        if (hi == 0)
            q[(size_t)bi * DD + wcol] = s_all[(size_t)bi * DD + wcol] + acc_dq;
    } else {
        acc0 += __shfl_xor(acc0, 16, 32);
        acc1 += __shfl_xor(acc1, 16, 32);
        acc2 += __shfl_xor(acc2, 16, 32);
        if (hi == 0) {
            float* o = ((G == 1) ? dmu1 : dmu2) + (size_t)bi * 3 * DD + wcol;
            o[0 * DD] = acc0; o[1 * DD] = acc1; o[2 * DD] = acc2;
        }
    }
}

// ---------------------------------------------------------------------------
// Kernel 3: mixing. One block per (b,i), 128 threads.
// mu = v_all + dmu1 + dmu2; mu@W_mu; rmsnorm(norm2); GLU chain; outputs.
// ---------------------------------------------------------------------------
__global__ __launch_bounds__(128)
void painn_mixing_kernel(const float* __restrict__ v_all,
                         const float* __restrict__ q,
                         const float* __restrict__ dmu1,
                         const float* __restrict__ dmu2,
                         const float* __restrict__ norm2_w,
                         const float* __restrict__ W_mu,
                         const float* __restrict__ W_m1, const float* __restrict__ b_m1,
                         const float* __restrict__ W_m2, const float* __restrict__ b_m2,
                         float* __restrict__ q_out,
                         float* __restrict__ mu_out)
{
    const int bi = blockIdx.x;
    const int b  = bi / NN, i = bi % NN;
    const int d  = threadIdx.x;

    __shared__ float sh_mu [3 * DD];
    __shared__ float sh_muV[3 * DD];
    __shared__ float sh_muW[3 * DD];
    __shared__ float sh_ctx[2 * DD];
    __shared__ float sh_h  [DD];
    __shared__ float red   [DD];

    const float* va = v_all + (size_t)bi * 3 * DD;
    const float* m1 = dmu1  + (size_t)bi * 3 * DD;
    const float* m2 = dmu2  + (size_t)bi * 3 * DD;
    #pragma unroll
    for (int r = 0; r < 3; ++r)
        sh_mu[r * DD + d] = va[r * DD + d] + m1[r * DD + d] + m2[r * DD + d];
    __syncthreads();

    // mu_mix = mu @ W_mu : V = cols [0,128), W = cols [128,256)
    #pragma unroll
    for (int r = 0; r < 3; ++r) {
        float v0 = 0.0f, v1 = 0.0f;
        #pragma unroll 4
        for (int k = 0; k < DD; ++k) {
            const float m = sh_mu[r * DD + k];
            v0 = fmaf(m, W_mu[k * 2 * DD + d],      v0);
            v1 = fmaf(m, W_mu[k * 2 * DD + d + DD], v1);
        }
        sh_muV[r * DD + d] = v0;
        sh_muW[r * DD + d] = v1;
    }

    const float qv = q[(size_t)bi * DD + d];
    red[d] = qv * qv;
    __syncthreads();
    for (int o = 64; o > 0; o >>= 1) { if (d < o) red[d] += red[d + o]; __syncthreads(); }
    const float var = red[0] * (1.0f / DD);
    const float qn  = qv * rsqrtf(var + 1e-6f) * (1.0f + norm2_w[d]);

    const float v0 = sh_muV[0 * DD + d], v1 = sh_muV[1 * DD + d], v2 = sh_muV[2 * DD + d];
    const float Vn = sqrtf(v0 * v0 + v1 * v1 + v2 * v2 + 1e-6f);
    sh_ctx[d] = qn; sh_ctx[d + DD] = Vn;
    __syncthreads();

    // x2 = GLU(ctx @ W_m1 + b_m1) @ W_m2 + b_m2
    float a0 = b_m1[d], g0 = b_m1[d + DD];
    #pragma unroll 4
    for (int k = 0; k < 2 * DD; ++k) {
        const float c = sh_ctx[k];
        a0 = fmaf(c, W_m1[k * 2 * DD + d],      a0);
        g0 = fmaf(c, W_m1[k * 2 * DD + d + DD], g0);
    }
    const float h = (a0 / (1.0f + expf(-a0))) * (1.0f / (1.0f + expf(-g0)));
    sh_h[d] = h;
    __syncthreads();

    float x0 = b_m2[d], x1 = b_m2[d + DD], x2v = b_m2[d + 2 * DD];
    #pragma unroll 4
    for (int k = 0; k < DD; ++k) {
        const float hh = sh_h[k];
        x0  = fmaf(hh, W_m2[k * TD + d],          x0);
        x1  = fmaf(hh, W_m2[k * TD + d + DD],     x1);
        x2v = fmaf(hh, W_m2[k * TD + d + 2 * DD], x2v);
    }

    const float dotVW = v0 * sh_muW[0 * DD + d] + v1 * sh_muW[1 * DD + d] + v2 * sh_muW[2 * DD + d];
    if (i < NH)
        q_out[((size_t)b * NH + i) * DD + d] = qv + x0 + x2v * dotVW;
    #pragma unroll
    for (int r = 0; r < 3; ++r)
        mu_out[(size_t)bi * 3 * DD + r * DD + d] = sh_mu[r * DD + d] + x1 * sh_muW[r * DD + d];
}

// ---------------------------------------------------------------------------
extern "C" void kernel_launch(void* const* d_in, const int* in_sizes, int n_in,
                              void* d_out, int out_size, void* d_ws, size_t ws_size,
                              hipStream_t stream) {
    const float* s_heavy  = (const float*)d_in[0];
    const float* v_all    = (const float*)d_in[1];
    const float* rbf      = (const float*)d_in[2];
    const float* dir_ij   = (const float*)d_in[3];
    const float* mask_ij  = (const float*)d_in[4];
    const int*   is_hyd   = (const int*)  d_in[5];
    const float* h_emb    = (const float*)d_in[6];
    const float* W_filter = (const float*)d_in[7];
    const float* b_filter = (const float*)d_in[8];
    const float* norm1_w  = (const float*)d_in[9];
    const float* W_i1     = (const float*)d_in[10];
    const float* b_i1     = (const float*)d_in[11];
    const float* W_i2     = (const float*)d_in[12];
    const float* b_i2     = (const float*)d_in[13];
    const float* norm2_w  = (const float*)d_in[14];
    const float* W_m1     = (const float*)d_in[15];
    const float* b_m1     = (const float*)d_in[16];
    const float* W_m2     = (const float*)d_in[17];
    const float* b_m2     = (const float*)d_in[18];
    const float* W_mu     = (const float*)d_in[19];

    // workspace layout (floats)
    float* ws     = (float*)d_ws;
    float* s_all  = ws;                                   // B*N*D       = 65536
    float* x      = s_all + (size_t)BB * NN * DD;         // B*N*3D      = 196608
    float* qbuf   = x     + (size_t)BB * NN * TD;         // B*N*D       = 65536
    float* dmu1   = qbuf  + (size_t)BB * NN * DD;         // B*N*3*D     = 196608
    float* dmu2   = dmu1  + (size_t)BB * NN * 3 * DD;     // B*N*3*D     = 196608

    float* q_out  = (float*)d_out;                        // (B,Nh,D)
    float* mu_out = q_out + (size_t)BB * NH * DD;         // (B,N,3,D)

    painn_prep_kernel<<<BB * NN, 128, 0, stream>>>(
        s_heavy, is_hyd, h_emb, norm1_w, W_i1, b_i1, W_i2, b_i2, s_all, x);

    painn_interaction_kernel<0><<<BB * NN, 256, 0, stream>>>(
        rbf, dir_ij, mask_ij, v_all, W_filter, b_filter, x, s_all, qbuf, dmu1, dmu2);
    painn_interaction_kernel<1><<<BB * NN, 256, 0, stream>>>(
        rbf, dir_ij, mask_ij, v_all, W_filter, b_filter, x, s_all, qbuf, dmu1, dmu2);
    painn_interaction_kernel<2><<<BB * NN, 256, 0, stream>>>(
        rbf, dir_ij, mask_ij, v_all, W_filter, b_filter, x, s_all, qbuf, dmu1, dmu2);

    painn_mixing_kernel<<<BB * NN, 128, 0, stream>>>(
        v_all, qbuf, dmu1, dmu2, norm2_w, W_mu, W_m1, b_m1, W_m2, b_m2,
        q_out, mu_out);
}
